// KNNMambaClassifier_72791105732716
// MI455X (gfx1250) — matrix operand order repaired and checked by reading, not compiled, not run on hardware
//
#include <hip/hip_runtime.h>
#include <hip/hip_bf16.h>
#include <math.h>

typedef __attribute__((ext_vector_type(16))) _Float16 v16h;
typedef __attribute__((ext_vector_type(8)))  _Float16 v8h;
typedef __attribute__((ext_vector_type(8)))  float    v8f;

#define TB        16            // batch elements per block
#define ROWS      (TB*6)        // 96 rows (batch*time)
#define NTHREADS  256           // 8 waves (wave32)

// shared memory carve (bytes)
#define OFF_H    0              // 96*64  f32 = 24576  (residual stream)
#define OFF_A16  24576          // 96*128 f16 = 24576  (WMMA A staging)
#define OFF_XC   49152          // 96*128 f32 = 49152  (x branch; reused as out-proj result)
#define OFF_Z    98304          // 96*128 f32 = 49152  (gate branch)
#define OFF_DBC  147456         // 96*48  f32 = 18432  (dt|B|C; reused as head feat)
#define SM_BYTES 165888

__device__ __forceinline__ float silu_f(float x)     { return x / (1.0f + __expf(-x)); }
__device__ __forceinline__ float softplus_f(float x) { return (x > 20.f) ? x : log1pf(__expf(x)); }

// One 16x16 output tile, D = A(16xK) * B(KxN-col-tile) via v_wmma_f32_16x16x32_f16.
// A16: LDS, row-major [rows][K] f16 (stride == K).
// W16: global, row-major [ncols][K] f16 (i.e. B^T == the weight matrix itself).
// Fragment layouts follow CDNA5 ISA 7.12.2:
//   A lane m=lane&15 holds K {kb+kh..kb+kh+7, kb+16+kh..kb+16+kh+7}, kh = (lane<16?0:8)
//   B lane n=lane&15 holds K {kb+kh..kb+kh+7, kb+16+kh..kb+16+kh+7} (same as A)
//   D vgpr r -> (M = r + (lane<16?0:8), N = lane&15)
template<int K>
__device__ __forceinline__ void gemm_tile16(const _Float16* __restrict__ A16,
                                            const _Float16* __restrict__ W16,
                                            int ncols, int mi, int wcolbase,
                                            float* __restrict__ C, int cstride, int ccolbase,
                                            int lane)
{
    v8f c = {};
    const int  l15   = lane & 15;
    const int  khalf = (lane < 16) ? 0 : 8;
    const int  bko   = (lane < 16) ? 0 : 16;
    const int  arow  = mi * 16 + l15;
    const int  wcol  = wcolbase + l15;
    const bool colok = (wcol < ncols);
#pragma unroll
    for (int ks = 0; ks < K / 32; ++ks) {
        const int kb = ks * 32;
        v8h alo = *(const v8h*)(A16 + (size_t)arow * K + kb + khalf);
        v8h ahi = *(const v8h*)(A16 + (size_t)arow * K + kb + 16 + khalf);
        v16h a;
#pragma unroll
        for (int i = 0; i < 8; ++i) { a[i] = alo[i]; a[i + 8] = ahi[i]; }
        v16h b = {};
        if (colok) {
            v8h blo = *(const v8h*)(W16 + (size_t)wcol * K + kb + khalf);
            v8h bhi = *(const v8h*)(W16 + (size_t)wcol * K + kb + 16 + khalf);
            for (int i = 0; i < 8; ++i) { b[i] = blo[i]; b[i + 8] = bhi[i]; }
        }
        c = __builtin_amdgcn_wmma_f32_16x16x32_f16(false, a, false, b, (short)0, c, false, false);
    }
    const int crow = mi * 16 + ((lane < 16) ? 0 : 8);
    const int ccol = ccolbase + l15;
#pragma unroll
    for (int r = 0; r < 8; ++r)
        C[(size_t)(crow + r) * cstride + ccol] = c[r];
}

// Convert the three GEMM weight tensors to f16 (layout preserved, row-major [out][in])
__global__ void prep_weights(const float* __restrict__ in_w,   // 2*256*64
                             const float* __restrict__ xp_w,   // 2*36*128
                             const float* __restrict__ op_w,   // 2*64*128
                             _Float16* __restrict__ w16)
{
    int i = blockIdx.x * blockDim.x + threadIdx.x;
    if (i < 32768)       w16[i] = (_Float16)in_w[i];
    else if (i < 41984)  w16[i] = (_Float16)xp_w[i - 32768];
    else if (i < 58368)  w16[i] = (_Float16)op_w[i - 41984];
}

__global__ __launch_bounds__(NTHREADS) void mamba_fused(
    const float* __restrict__ x,
    const float* __restrict__ proj_w, const float* __restrict__ proj_b,
    const float* __restrict__ conv_w, const float* __restrict__ conv_b,
    const float* __restrict__ dt_w,   const float* __restrict__ dt_b,
    const float* __restrict__ A_log,  const float* __restrict__ Dp,
    const float* __restrict__ ln_g,   const float* __restrict__ ln_b,
    const float* __restrict__ h_w1,   const float* __restrict__ h_b1,
    const float* __restrict__ h_w2,   const float* __restrict__ h_b2,
    const _Float16* __restrict__ w16in, const _Float16* __restrict__ w16xp,
    const _Float16* __restrict__ w16op,
    float* __restrict__ out)
{
    __shared__ __attribute__((aligned(64))) char smem[SM_BYTES];
    float*    sm_h   = (float*)(smem + OFF_H);
    _Float16* sm_a16 = (_Float16*)(smem + OFF_A16);
    float*    sm_xc  = (float*)(smem + OFF_XC);
    float*    sm_z   = (float*)(smem + OFF_Z);
    float*    sm_dbc = (float*)(smem + OFF_DBC);
    float*    sm_out = sm_xc;   // out-proj result reuses xc region (dead by then)

    const int tid  = threadIdx.x;
    const int lane = tid & 31;
    const int wv   = tid >> 5;
    const int b0   = blockIdx.x * TB;

    // ---- phase 0: h = xs @ proj_w.T + proj_b (K=4: plain FMA) ----
    for (int i = tid; i < ROWS * 64; i += NTHREADS) {
        int row = i >> 6, d = i & 63;
        const float* xr = x + ((size_t)(b0 + row / 6) * 6 + (row % 6)) * 8;
        const float* pw = proj_w + d * 4;
        sm_h[i] = proj_b[d] + xr[0]*pw[0] + xr[1]*pw[1] + xr[2]*pw[2] + xr[3]*pw[3];
    }
    __syncthreads();

    for (int L = 0; L < 2; ++L) {
        const _Float16* Win = w16in + L * 256 * 64;
        const _Float16* Wxp = w16xp + L * 36 * 128;
        const _Float16* Wop = w16op + L * 64 * 128;
        const float* cw  = conv_w + L * 128 * 4;
        const float* cb  = conv_b + L * 128;
        const float* dtw = dt_w  + L * 128 * 4;
        const float* dtb = dt_b  + L * 128;
        const float* Al  = A_log + L * 128 * 16;
        const float* Dpl = Dp    + L * 128;
        const float* lg  = ln_g  + L * 64;
        const float* lb  = ln_b  + L * 64;

        // ---- phase 1: stage residual stream h -> f16 A operand (stride 64) ----
        for (int i = tid; i < ROWS * 64; i += NTHREADS)
            sm_a16[i] = (_Float16)sm_h[i];
        __syncthreads();

        // ---- phase 2: GEMM1 xz = h @ in_w.T  (96 x 64 x 256) ----
        for (int tix = wv; tix < 6 * 16; tix += 8) {
            int mi = tix % 6, ni = tix / 6;
            float* C  = (ni < 8) ? sm_xc : sm_z;
            int   ccb = (ni & 7) * 16;
            gemm_tile16<64>(sm_a16, Win, 256, mi, ni * 16, C, 128, ccb, lane);
        }
        __syncthreads();

        // ---- phase 3: depthwise causal conv (k=4) + SiLU; stage xc as f32 + f16 ----
        for (int p = tid; p < TB * 128; p += NTHREADS) {
            int b = p >> 7, ch = p & 127;
            float w0 = cw[ch*4+0], w1 = cw[ch*4+1], w2 = cw[ch*4+2], w3 = cw[ch*4+3];
            float bias = cb[ch];
            float in[6];
#pragma unroll
            for (int t = 0; t < 6; ++t) in[t] = sm_xc[(size_t)(b*6 + t) * 128 + ch];
#pragma unroll
            for (int t = 0; t < 6; ++t) {
                float v = bias + w3 * in[t];
                if (t >= 1) v += w2 * in[t-1];
                if (t >= 2) v += w1 * in[t-2];
                if (t >= 3) v += w0 * in[t-3];
                float s = silu_f(v);
                sm_xc[(size_t)(b*6 + t) * 128 + ch]  = s;
                sm_a16[(size_t)(b*6 + t) * 128 + ch] = (_Float16)s;
            }
        }
        __syncthreads();

        // ---- phase 4: GEMM2 dbc = xc @ xp_w.T (96 x 128 x 36, padded to 48 cols) ----
        for (int tix = wv; tix < 6 * 3; tix += 8) {
            int mi = tix % 6, ni = tix / 6;
            gemm_tile16<128>(sm_a16, Wxp, 36, mi, ni * 16, sm_dbc, 48, ni * 16, lane);
        }
        __syncthreads();

        // ---- phase 5: selective scan (dt computed on the fly), gate, stage y f16 ----
        for (int p = tid; p < TB * 128; p += NTHREADS) {
            int b = p >> 7, ch = p & 127;
            float d0 = dtw[ch*4+0], d1 = dtw[ch*4+1], d2 = dtw[ch*4+2], d3 = dtw[ch*4+3];
            float db = dtb[ch];
            float Dv = Dpl[ch];
            float Av[16], hst[16];
#pragma unroll
            for (int s = 0; s < 16; ++s) { Av[s] = -__expf(Al[ch*16 + s]); hst[s] = 0.f; }
            for (int t = 0; t < 6; ++t) {
                const float* r = sm_dbc + (size_t)(b*6 + t) * 48;
                float dt  = softplus_f(r[0]*d0 + r[1]*d1 + r[2]*d2 + r[3]*d3 + db);
                float xcv = sm_xc[(size_t)(b*6 + t) * 128 + ch];
                float y = 0.f;
#pragma unroll
                for (int s = 0; s < 16; ++s) {
                    float dA = __expf(dt * Av[s]);
                    hst[s] = dA * hst[s] + (dt * r[4 + s]) * xcv;
                    y += hst[s] * r[20 + s];
                }
                float zv = sm_z[(size_t)(b*6 + t) * 128 + ch];
                float yg = (y + xcv * Dv) * silu_f(zv);
                sm_a16[(size_t)(b*6 + t) * 128 + ch] = (_Float16)yg;
            }
        }
        __syncthreads();

        // ---- phase 6: GEMM3 o = y @ out_w.T (96 x 128 x 64) -> sm_out ----
        for (int tix = wv; tix < 6 * 4; tix += 8) {
            int mi = tix % 6, ni = tix / 6;
            gemm_tile16<128>(sm_a16, Wop, 64, mi, ni * 16, sm_out, 64, ni * 16, lane);
        }
        __syncthreads();

        // ---- phase 7: layernorm + residual into h ----
        for (int row = tid; row < ROWS; row += NTHREADS) {
            const float* o = sm_out + (size_t)row * 64;
            float mu = 0.f;
            for (int d = 0; d < 64; ++d) mu += o[d];
            mu *= (1.f / 64.f);
            float var = 0.f;
            for (int d = 0; d < 64; ++d) { float dd = o[d] - mu; var += dd * dd; }
            var *= (1.f / 64.f);
            float rs = rsqrtf(var + 1e-5f);
            for (int d = 0; d < 64; ++d)
                sm_h[(size_t)row * 64 + d] += (o[d] - mu) * rs * lg[d] + lb[d];
        }
        __syncthreads();
    }

    // ---- head: feat = [h[:,0], mean(h[:,1:]), max(h[:,1:])] -> relu MLP -> sigmoid ----
    float* sm_feat = sm_dbc;                    // 16*192 f32 fits in dbc region
    for (int i = tid; i < TB * 64; i += NTHREADS) {
        int b = i >> 6, d = i & 63;
        float pr = sm_h[(size_t)(b*6) * 64 + d];
        float mn = 0.f, mx = -3.402823466e38f;
#pragma unroll
        for (int t = 1; t < 6; ++t) {
            float v = sm_h[(size_t)(b*6 + t) * 64 + d];
            mn += v; mx = fmaxf(mx, v);
        }
        sm_feat[b*192 + d]       = pr;
        sm_feat[b*192 + 64 + d]  = mn * 0.2f;
        sm_feat[b*192 + 128 + d] = mx;
    }
    __syncthreads();
    float* sm_z1 = (float*)(smem + OFF_A16);    // 16*32 f32, a16 region is dead
    for (int i = tid; i < TB * 32; i += NTHREADS) {
        int b = i >> 5, j = i & 31;
        float acc = h_b1[j];
        const float* wr = h_w1 + (size_t)j * 192;
        const float* fr = sm_feat + (size_t)b * 192;
        for (int k = 0; k < 192; ++k) acc += fr[k] * wr[k];
        sm_z1[b*32 + j] = fmaxf(acc, 0.f);
    }
    __syncthreads();
    for (int b = tid; b < TB; b += NTHREADS) {
        float acc = h_b2[0];
        for (int j = 0; j < 32; ++j) acc += sm_z1[b*32 + j] * h_w2[j];
        out[b0 + b] = 1.f / (1.f + __expf(-acc));
    }
}

extern "C" void kernel_launch(void* const* d_in, const int* in_sizes, int n_in,
                              void* d_out, int out_size, void* d_ws, size_t ws_size,
                              hipStream_t stream)
{
    (void)in_sizes; (void)n_in; (void)out_size; (void)ws_size;
    const float* x      = (const float*)d_in[0];
    const float* proj_w = (const float*)d_in[1];
    const float* proj_b = (const float*)d_in[2];
    const float* in_w   = (const float*)d_in[3];
    const float* conv_w = (const float*)d_in[4];
    const float* conv_b = (const float*)d_in[5];
    const float* xp_w   = (const float*)d_in[6];
    const float* dt_w   = (const float*)d_in[7];
    const float* dt_b   = (const float*)d_in[8];
    const float* A_log  = (const float*)d_in[9];
    const float* Dp     = (const float*)d_in[10];
    const float* op_w   = (const float*)d_in[11];
    const float* ln_g   = (const float*)d_in[12];
    const float* ln_b   = (const float*)d_in[13];
    const float* hw1    = (const float*)d_in[14];
    const float* hb1    = (const float*)d_in[15];
    const float* hw2    = (const float*)d_in[16];
    const float* hb2    = (const float*)d_in[17];

    _Float16* w16 = (_Float16*)d_ws;
    prep_weights<<<(58368 + 255) / 256, 256, 0, stream>>>(in_w, xp_w, op_w, w16);

    const _Float16* w16in = w16;
    const _Float16* w16xp = w16 + 32768;   // 2*256*64
    const _Float16* w16op = w16 + 41984;   // + 2*36*128

    mamba_fused<<<4096 / TB, NTHREADS, 0, stream>>>(
        x, proj_w, proj_b, conv_w, conv_b, dt_w, dt_b, A_log, Dp,
        ln_g, ln_b, hw1, hb1, hw2, hb2, w16in, w16xp, w16op, (float*)d_out);
}
